// InvariantSlotAttention_74397423501774
// MI455X (gfx1250) — matrix operand order repaired and checked by reading, not compiled, not run on hardware
//
#include <hip/hip_runtime.h>
#include <stdint.h>

// ---------------------------------------------------------------------------
// Invariant Slot Attention, fused for MI455X (gfx1250, wave32, WMMA).
// Heavy matmuls: v_wmma_f32_16x16x32_f16 (f16 in, f32 accumulate).
// LDS operands stored in *fragment order*: each lane's 16-half fragment is
// two 16B vectors -> ds_load_b128.  All staging/build loops use 128-bit ops.
// ---------------------------------------------------------------------------

typedef __attribute__((ext_vector_type(16))) _Float16 v16h;
typedef __attribute__((ext_vector_type(8)))  float    v8f;
typedef __attribute__((ext_vector_type(4)))  uint32_t v4u;
typedef __attribute__((ext_vector_type(4)))  float    v4f;

union Frag16 { v16h h; v4u q[2]; };
union Pack8h { v4u q; _Float16 hh[8]; };

#define WMMA_F16(A, B, C) \
  __builtin_amdgcn_wmma_f32_16x16x32_f16(false, (A), false, (B), (short)0, (C), false, false)

#define ZERO8 {0.f,0.f,0.f,0.f,0.f,0.f,0.f,0.f}

// ---- fragment-order LDS layout --------------------------------------------
// A-frag (16-bit A 16x32, ISA 7.12.2): lane L holds row L&15,
// kbase = kchunk*32 + (L>>4)*8, pair p -> K = kbase + (p<4 ? 2p : 16+2(p-4)).
// Tile stored so frag kchunk / lane L = v4u slots [kchunk*64 + L*2, +1].
__device__ inline v16h load_A_sw(const _Float16* tile, int kchunk) {
  int lane = threadIdx.x & 31;
  const v4u* p = (const v4u*)tile + kchunk * 64 + lane * 2;
  Frag16 f;
  f.q[0] = p[0];
  f.q[1] = p[1];          // 2x ds_load_b128
  return f.h;
}

// B-frag: lane L holds col nn = nchunk*16 + (L&15), kb = kchunk*32 + (L>>4)*8,
// pair p -> K = kb + (p<4 ? 2p : 16+2(p-4)).  Stored so frag (nc,kc)/lane L = v4u slots
// [((nc*4+kc)*32 + L)*2, +1].
__device__ inline v16h load_B_sw(const _Float16* w, int nchunk, int kchunk) {
  int lane = threadIdx.x & 31;
  const v4u* p = (const v4u*)w + ((nchunk * 4 + kchunk) * 32 + lane) * 2;
  Frag16 f;
  f.q[0] = p[0];
  f.q[1] = p[1];
  return f.h;
}

// Stage 128x128 f16 weights (row-major W[N][K]) into B-fragment order.
// The permutation maps 4 consecutive src dwords to 4 consecutive dst dwords,
// so the whole copy is b128 global loads -> b128 LDS stores.
__device__ inline void stage_B_sw(v4u* dst, const v4u* src) {
  int tid = threadIdx.x;
  for (int c = tid; c < 2048; c += 256) {
    int half = c & 1;
    int L = (c >> 1) & 31;
    int f = c >> 6;
    int kc = f & 3, nc = f >> 2;
    int nn = nc * 16 + (L & 15);
    dst[c] = src[nn * 16 + kc * 4 + (L >> 4) + half * 2];
  }
}

// ---- 2-layer MLP, layer 1 (x @ w1.T + b1, relu) -> hidden A-frags ----------
__device__ inline void mlp_layer1(const _Float16* xw, _Float16* hw,
                                  const _Float16* w1l, const float* fb1,
                                  v16h hfr[4]) {
  v16h a0 = load_A_sw(xw, 0), a1 = load_A_sw(xw, 1);
  v16h a2 = load_A_sw(xw, 2), a3 = load_A_sw(xw, 3);
  int lane = threadIdx.x & 31;
  int lm = lane & 15, mb = ((lane >> 4) & 1) * 8;
#pragma unroll
  for (int nc = 0; nc < 8; ++nc) {
    v8f acc = ZERO8;
    acc = WMMA_F16(a0, load_B_sw(w1l, nc, 0), acc);
    acc = WMMA_F16(a1, load_B_sw(w1l, nc, 1), acc);
    acc = WMMA_F16(a2, load_B_sw(w1l, nc, 2), acc);
    acc = WMMA_F16(a3, load_B_sw(w1l, nc, 3), acc);
    int nn = nc * 16 + lm;
    float bb = fb1[nn];
    // scatter C-frag element (row M = mb+r, feature nn) into A-frag slot
    int kc = nn >> 5, kk = nn & 31;
    int h = (kk >> 3) & 1, o = kk - h * 8;
    int p = (o < 8) ? (o >> 1) : (4 + ((o - 16) >> 1));
    int base = (kc * 256 + (h * 16 + mb) * 8 + p) * 2 + (o & 1);
#pragma unroll
    for (int r = 0; r < 8; ++r) {
      float hv = acc[r] + bb;
      hv = hv > 0.f ? hv : 0.f;
      hw[base + r * 16] = (_Float16)hv;     // LDS in-order within wave
    }
  }
#pragma unroll
  for (int c = 0; c < 4; ++c) hfr[c] = load_A_sw(hw, c);
}

// ---- elementwise prep ------------------------------------------------------
__global__ void cvt_f32_f16_kernel(const float* __restrict__ s,
                                   _Float16* __restrict__ d, int n) {
  int i = blockIdx.x * 256 + threadIdx.x;
  if (i < n) d[i] = (_Float16)s[i];
}

__global__ void copy_f32_kernel(const float* __restrict__ s,
                                float* __restrict__ d, int n) {
  int i = blockIdx.x * 256 + threadIdx.x;
  if (i < n) d[i] = s[i];
}

// ---- to_k / to_v = inputs @ Wk.T / Wv.T  (f16 out, WMMA) -------------------
__global__ void __launch_bounds__(256)
tokv_kernel(const _Float16* __restrict__ in16,
            const _Float16* __restrict__ wkg, const _Float16* __restrict__ wvg,
            _Float16* __restrict__ tok, _Float16* __restrict__ tov) {
  extern __shared__ char smem[];
  _Float16* wkl = (_Float16*)smem;      // B-swizzled
  _Float16* wvl = wkl + 16384;
  _Float16* xt  = wvl + 16384;          // 8 waves * 2048 halves, A-swizzled
  int tid = threadIdx.x, wave = tid >> 5, lane = tid & 31;
  stage_B_sw((v4u*)wkl, (const v4u*)wkg);
  stage_B_sw((v4u*)wvl, (const v4u*)wvg);
  __syncthreads();

  size_t rowbase = (size_t)blockIdx.x * 128 + (size_t)wave * 16;  // of 32768 rows
  _Float16* xw = xt + wave * 2048;
  v4u* xw4 = (v4u*)xw;
  const v4u* in4 = (const v4u*)(in16 + rowbase * 128);

  // stage tile in A-fragment order, pure b128 copy
  int g = lane & 15, k0 = g * 8;
  int kc0 = k0 >> 5, kk0 = k0 & 31;
  int hh = (kk0 >> 3) & 1, pq = (kk0 & 16) ? 1 : 0;
#pragma unroll
  for (int t = 0; t < 8; ++t) {
    int row = (lane >> 4) + 2 * t;
    xw4[kc0 * 64 + (hh * 16 + row) * 2 + pq] = in4[row * 16 + g];
  }

  v16h a0 = load_A_sw(xw, 0), a1 = load_A_sw(xw, 1);
  v16h a2 = load_A_sw(xw, 2), a3 = load_A_sw(xw, 3);
  int lm = lane & 15, mb = ((lane >> 4) & 1) * 8;
#pragma unroll
  for (int nc = 0; nc < 8; ++nc) {
    v8f ak = ZERO8, av = ZERO8;
    ak = WMMA_F16(a0, load_B_sw(wkl, nc, 0), ak);
    ak = WMMA_F16(a1, load_B_sw(wkl, nc, 1), ak);
    ak = WMMA_F16(a2, load_B_sw(wkl, nc, 2), ak);
    ak = WMMA_F16(a3, load_B_sw(wkl, nc, 3), ak);
    av = WMMA_F16(a0, load_B_sw(wvl, nc, 0), av);
    av = WMMA_F16(a1, load_B_sw(wvl, nc, 1), av);
    av = WMMA_F16(a2, load_B_sw(wvl, nc, 2), av);
    av = WMMA_F16(a3, load_B_sw(wvl, nc, 3), av);
    int nn = nc * 16 + lm;
#pragma unroll
    for (int r = 0; r < 8; ++r) {
      tok[(rowbase + mb + r) * 128 + nn] = (_Float16)ak[r];
      tov[(rowbase + mb + r) * 128 + nn] = (_Float16)av[r];
    }
  }
}

// ---- LN(slots) @ Wq.T ------------------------------------------------------
__global__ void __launch_bounds__(128)
lnq_kernel(const float* __restrict__ slots, const float* __restrict__ g,
           const float* __restrict__ bta, const float* __restrict__ Wq,
           float* __restrict__ q) {
  __shared__ float xn[128], red[128];
  int row = blockIdx.x, t = threadIdx.x;
  float v = slots[row * 128 + t];
  red[t] = v; __syncthreads();
  for (int s = 64; s > 0; s >>= 1) { if (t < s) red[t] += red[t + s]; __syncthreads(); }
  float mean = red[0] * (1.f / 128.f); __syncthreads();
  float d = v - mean;
  red[t] = d * d; __syncthreads();
  for (int s = 64; s > 0; s >>= 1) { if (t < s) red[t] += red[t + s]; __syncthreads(); }
  float var = red[0] * (1.f / 128.f); __syncthreads();
  xn[t] = d * rsqrtf(var + 1e-5f) * g[t] + bta[t];
  __syncthreads();
  float acc = 0.f;
  const float* wr = Wq + t * 128;
  for (int c = 0; c < 128; ++c) acc += xn[c] * wr[c];
  q[row * 128 + t] = acc;
}

// ---- dots = (mlp2(to_k + pos) . q) * scale  (WMMA-fused) -------------------
__global__ void __launch_bounds__(256)
dots_kernel(const _Float16* __restrict__ tok,
            const _Float16* __restrict__ w1g, const _Float16* __restrict__ w2g,
            const float* __restrict__ fb1, const float* __restrict__ fb2,
            const float* __restrict__ gw, const float* __restrict__ gb,
            const float* __restrict__ spcur, const float* __restrict__ qb,
            float* __restrict__ dotsb) {
  extern __shared__ char smem[];
  _Float16* w1l = (_Float16*)smem;
  _Float16* w2l = w1l + 16384;
  _Float16* xt  = w2l + 16384;
  _Float16* ht  = xt + 16384;
  int tid = threadIdx.x, wave = tid >> 5, lane = tid & 31;
  int s = blockIdx.y, b = blockIdx.z, bs = b * 8 + s;
  stage_B_sw((v4u*)w1l, (const v4u*)w1g);
  stage_B_sw((v4u*)w2l, (const v4u*)w2g);
  __syncthreads();

  // per-lane pos coefficients: lane's 8 feature columns are fixed
  int g = lane & 15, k0 = g * 8;
  float cyr[8], cxr[8], cbr[8];
#pragma unroll
  for (int e = 0; e < 8; ++e) {
    cyr[e] = gw[(k0 + e) * 2];
    cxr[e] = gw[(k0 + e) * 2 + 1];
    cbr[e] = gb[k0 + e];
  }
  int kc0 = k0 >> 5, kk0 = k0 & 31;
  int hh = (kk0 >> 3) & 1, pq = (kk0 & 16) ? 1 : 0;

  int rowbase = blockIdx.x * 128 + wave * 16;         // pixel base
  float sy = spcur[bs * 2 + 0], sx = spcur[bs * 2 + 1];
  _Float16* xw = xt + wave * 2048;
  _Float16* hw = ht + wave * 2048;
  v4u* xw4 = (v4u*)xw;
  const v4u* tk4 = (const v4u*)(tok + ((size_t)b * 4096 + rowbase) * 128);

  // x = to_k + pos(rel), built directly in A-fragment order (b128 in/out)
#pragma unroll
  for (int t = 0; t < 8; ++t) {
    int row = (lane >> 4) + 2 * t;
    int j = rowbase + row;
    float gy = -1.f + (float)(j >> 6) * (2.f / 63.f);
    float gx = -1.f + (float)(j & 63) * (2.f / 63.f);
    float ry = (gy - sy) * 0.2f, rx = (gx - sx) * 0.2f;   // /DELTA
    Pack8h pk; pk.q = tk4[row * 16 + g];
    Pack8h po;
#pragma unroll
    for (int e = 0; e < 8; ++e)
      po.hh[e] = (_Float16)((float)pk.hh[e] + cyr[e] * ry + cxr[e] * rx + cbr[e]);
    xw4[kc0 * 64 + (hh * 16 + row) * 2 + pq] = po.q;
  }

  v16h hfr[4];
  mlp_layer1(xw, hw, w1l, fb1, hfr);

  const float* qrow = qb + bs * 128;
  int lm = lane & 15, mb = ((lane >> 4) & 1) * 8;
  float rowsum[8] = ZERO8;
#pragma unroll
  for (int nc = 0; nc < 8; ++nc) {
    v8f acc = ZERO8;
    acc = WMMA_F16(hfr[0], load_B_sw(w2l, nc, 0), acc);
    acc = WMMA_F16(hfr[1], load_B_sw(w2l, nc, 1), acc);
    acc = WMMA_F16(hfr[2], load_B_sw(w2l, nc, 2), acc);
    acc = WMMA_F16(hfr[3], load_B_sw(w2l, nc, 3), acc);
    int nn = nc * 16 + lm;
    float bb = fb2[nn], qv = qrow[nn];
#pragma unroll
    for (int r = 0; r < 8; ++r) rowsum[r] += (acc[r] + bb) * qv;
  }
#pragma unroll
  for (int m = 1; m < 16; m <<= 1)
#pragma unroll
    for (int r = 0; r < 8; ++r) rowsum[r] += __shfl_xor(rowsum[r], m, 32);
  if (lm == 0) {
    const float scale = 0.08838834764831845f;  // 128^-0.5
#pragma unroll
    for (int r = 0; r < 8; ++r)
      dotsb[(size_t)bs * 4096 + rowbase + mb + r] = rowsum[r] * scale;
  }
}

// ---- softmax over slots (per pixel), + EPS ---------------------------------
__global__ void softmax_slots_kernel(float* __restrict__ dots) {
  int idx = blockIdx.x * 256 + threadIdx.x;   // b*n = 32768
  if (idx >= 32768) return;
  int b = idx >> 12, j = idx & 4095;
  float v[8], mx = -3.4e38f;
#pragma unroll
  for (int s = 0; s < 8; ++s) {
    v[s] = dots[((size_t)(b * 8 + s)) * 4096 + j];
    mx = fmaxf(mx, v[s]);
  }
  float sum = 0.f;
#pragma unroll
  for (int s = 0; s < 8; ++s) { v[s] = __expf(v[s] - mx); sum += v[s]; }
  float inv = 1.f / sum;
#pragma unroll
  for (int s = 0; s < 8; ++s)
    dots[((size_t)(b * 8 + s)) * 4096 + j] = v[s] * inv + 1e-8f;
}

// ---- per-slot normalize + attention-weighted centroid ----------------------
__global__ void __launch_bounds__(256)
attn_norm_kernel(float* __restrict__ attn, float* __restrict__ sp_next) {
  int s = blockIdx.x, b = blockIdx.y, bs = b * 8 + s;
  float* p = attn + (size_t)bs * 4096;
  __shared__ float r0[256], r1[256], r2[256];
  int t = threadIdx.x;
  float a0 = 0.f, a1 = 0.f, a2 = 0.f;
  for (int j = t; j < 4096; j += 256) {
    float pv = p[j];
    float gy = -1.f + (float)(j >> 6) * (2.f / 63.f);
    float gx = -1.f + (float)(j & 63) * (2.f / 63.f);
    a0 += pv; a1 += pv * gy; a2 += pv * gx;
  }
  r0[t] = a0; r1[t] = a1; r2[t] = a2; __syncthreads();
  for (int sft = 128; sft > 0; sft >>= 1) {
    if (t < sft) { r0[t] += r0[t + sft]; r1[t] += r1[t + sft]; r2[t] += r2[t + sft]; }
    __syncthreads();
  }
  float inv = 1.f / r0[0];
  if (t == 0) {
    sp_next[bs * 2 + 0] = r1[0] * inv;
    sp_next[bs * 2 + 1] = r2[0] * inv;
  }
  for (int j = t; j < 4096; j += 256) p[j] *= inv;
}

// ---- updates = sum_j attn_j * mlp2(to_v + pos)_j  (WMMA-fused, no atomics) -
__global__ void __launch_bounds__(256)
updates_kernel(const _Float16* __restrict__ tov,
               const _Float16* __restrict__ w1g, const _Float16* __restrict__ w2g,
               const float* __restrict__ fb1, const float* __restrict__ fb2,
               const float* __restrict__ gw, const float* __restrict__ gb,
               const float* __restrict__ spcur, const float* __restrict__ attn,
               float* __restrict__ updates) {
  extern __shared__ char smem[];
  _Float16* w1l = (_Float16*)smem;
  _Float16* w2l = w1l + 16384;
  _Float16* xt  = w2l + 16384;
  _Float16* ht  = xt + 16384;
  float* red    = (float*)(ht + 16384);   // 8 waves * 128
  int tid = threadIdx.x, wave = tid >> 5, lane = tid & 31;
  int s = blockIdx.x, b = blockIdx.y, bs = b * 8 + s;
  stage_B_sw((v4u*)w1l, (const v4u*)w1g);
  stage_B_sw((v4u*)w2l, (const v4u*)w2g);
  __syncthreads();

  int g = lane & 15, k0 = g * 8;
  float cyr[8], cxr[8], cbr[8];
#pragma unroll
  for (int e = 0; e < 8; ++e) {
    cyr[e] = gw[(k0 + e) * 2];
    cxr[e] = gw[(k0 + e) * 2 + 1];
    cbr[e] = gb[k0 + e];
  }
  int kc0 = k0 >> 5, kk0 = k0 & 31;
  int hh = (kk0 >> 3) & 1, pq = (kk0 & 16) ? 1 : 0;

  float sy = spcur[bs * 2 + 0], sx = spcur[bs * 2 + 1];
  _Float16* xw = xt + wave * 2048;
  _Float16* hw = ht + wave * 2048;
  v4u* xw4 = (v4u*)xw;
  const float* attnrow = attn + (size_t)bs * 4096;
  int lm = lane & 15, mb = ((lane >> 4) & 1) * 8;
  float colacc[8] = ZERO8;

  for (int it = 0; it < 32; ++it) {
    int rowbase = (it * 8 + wave) * 16;
    const v4u* tv4 = (const v4u*)(tov + ((size_t)b * 4096 + rowbase) * 128);
#pragma unroll
    for (int t = 0; t < 8; ++t) {
      int row = (lane >> 4) + 2 * t;
      int j = rowbase + row;
      float gy = -1.f + (float)(j >> 6) * (2.f / 63.f);
      float gx = -1.f + (float)(j & 63) * (2.f / 63.f);
      float ry = (gy - sy) * 0.2f, rx = (gx - sx) * 0.2f;
      Pack8h pk; pk.q = tv4[row * 16 + g];
      Pack8h po;
#pragma unroll
      for (int e = 0; e < 8; ++e)
        po.hh[e] = (_Float16)((float)pk.hh[e] + cyr[e] * ry + cxr[e] * rx + cbr[e]);
      xw4[kc0 * 64 + (hh * 16 + row) * 2 + pq] = po.q;
    }

    v16h hfr[4];
    mlp_layer1(xw, hw, w1l, fb1, hfr);

    const v4f* ap = (const v4f*)(attnrow + rowbase + mb);
    v4f aw0 = ap[0], aw1 = ap[1];           // 2x b128 loads
#pragma unroll
    for (int nc = 0; nc < 8; ++nc) {
      v8f acc = ZERO8;
      acc = WMMA_F16(hfr[0], load_B_sw(w2l, nc, 0), acc);
      acc = WMMA_F16(hfr[1], load_B_sw(w2l, nc, 1), acc);
      acc = WMMA_F16(hfr[2], load_B_sw(w2l, nc, 2), acc);
      acc = WMMA_F16(hfr[3], load_B_sw(w2l, nc, 3), acc);
      float bb = fb2[nc * 16 + lm];
      float part = 0.f;
#pragma unroll
      for (int r = 0; r < 4; ++r) part += (acc[r] + bb) * aw0[r];
#pragma unroll
      for (int r = 0; r < 4; ++r) part += (acc[r + 4] + bb) * aw1[r];
      colacc[nc] += part;
    }
  }
#pragma unroll
  for (int nc = 0; nc < 8; ++nc) colacc[nc] += __shfl_xor(colacc[nc], 16, 32);
  if (lane < 16) {
#pragma unroll
    for (int nc = 0; nc < 8; ++nc) red[wave * 128 + nc * 16 + lm] = colacc[nc];
  }
  __syncthreads();
  if (tid < 128) {
    float acc = 0.f;
    for (int w = 0; w < 8; ++w) acc += red[w * 128 + tid];
    updates[bs * 128 + tid] = acc;
  }
}

// ---- GRU + residual MLP slot update ----------------------------------------
__global__ void __launch_bounds__(128)
slot_update_kernel(const float* __restrict__ updates, float* __restrict__ slots,
                   const float* __restrict__ wih, const float* __restrict__ whh,
                   const float* __restrict__ bih, const float* __restrict__ bhh,
                   const float* __restrict__ mw1, const float* __restrict__ mb1,
                   const float* __restrict__ mw2, const float* __restrict__ mb2,
                   const float* __restrict__ lng, const float* __restrict__ lnb) {
  __shared__ float u[128], h[128], xln[128], hid[128], red[128];
  int row = blockIdx.x, t = threadIdx.x;
  u[t] = updates[row * 128 + t];
  h[t] = slots[row * 128 + t];
  __syncthreads();
  float gi_r = bih[t], gi_z = bih[128 + t], gi_n = bih[256 + t];
  float gh_r = bhh[t], gh_z = bhh[128 + t], gh_n = bhh[256 + t];
  const float* wr = wih + t * 128;
  const float* wz = wih + (128 + t) * 128;
  const float* wn = wih + (256 + t) * 128;
  const float* vr = whh + t * 128;
  const float* vz = whh + (128 + t) * 128;
  const float* vn = whh + (256 + t) * 128;
  for (int c = 0; c < 128; ++c) {
    float uc = u[c], hc = h[c];
    gi_r += wr[c] * uc; gi_z += wz[c] * uc; gi_n += wn[c] * uc;
    gh_r += vr[c] * hc; gh_z += vz[c] * hc; gh_n += vn[c] * hc;
  }
  float r = 1.f / (1.f + __expf(-(gi_r + gh_r)));
  float z = 1.f / (1.f + __expf(-(gi_z + gh_z)));
  float nn = tanhf(gi_n + r * gh_n);
  float hn = (1.f - z) * nn + z * h[t];
  // LayerNorm(hn)
  red[t] = hn; __syncthreads();
  for (int sft = 64; sft > 0; sft >>= 1) { if (t < sft) red[t] += red[t + sft]; __syncthreads(); }
  float mean = red[0] * (1.f / 128.f); __syncthreads();
  float d = hn - mean;
  red[t] = d * d; __syncthreads();
  for (int sft = 64; sft > 0; sft >>= 1) { if (t < sft) red[t] += red[t + sft]; __syncthreads(); }
  float var = red[0] * (1.f / 128.f); __syncthreads();
  xln[t] = d * rsqrtf(var + 1e-5f) * lng[t] + lnb[t];
  __syncthreads();
  float a = mb1[t];
  const float* m1 = mw1 + t * 128;
  for (int c = 0; c < 128; ++c) a += m1[c] * xln[c];
  hid[t] = a > 0.f ? a : 0.f;
  __syncthreads();
  float o = mb2[t];
  const float* m2 = mw2 + t * 128;
  for (int c = 0; c < 128; ++c) o += m2[c] * hid[c];
  slots[row * 128 + t] = hn + o;
}

// ---- pack outputs: (slots, S_p, S_r, S_s) ----------------------------------
__global__ void pack_out_kernel(const float* __restrict__ slots,
                                const float* __restrict__ sp,
                                const float* __restrict__ Sr,
                                const float* __restrict__ Ss,
                                float* __restrict__ out) {
  int i = blockIdx.x * 256 + threadIdx.x;
  if (i < 8192)       out[i] = slots[i];
  else if (i < 8320)  out[i] = sp[i - 8192];
  else if (i < 8576)  out[i] = Sr[i - 8320];
  else if (i < 8704)  out[i] = Ss[i - 8576];
}

// ---------------------------------------------------------------------------
extern "C" void kernel_launch(void* const* d_in, const int* in_sizes, int n_in,
                              void* d_out, int out_size, void* d_ws, size_t ws_size,
                              hipStream_t stream) {
  (void)in_sizes; (void)n_in; (void)out_size; (void)ws_size;
  const float* inputs     = (const float*)d_in[0];
  const float* slots_init = (const float*)d_in[1];
  const float* S_p0       = (const float*)d_in[2];
  const float* S_s        = (const float*)d_in[3];
  const float* S_r        = (const float*)d_in[4];
  const float* Wq         = (const float*)d_in[5];
  const float* Wk         = (const float*)d_in[6];
  const float* Wv         = (const float*)d_in[7];
  const float* g_w        = (const float*)d_in[8];
  const float* g_b        = (const float*)d_in[9];
  const float* f_w1       = (const float*)d_in[10];
  const float* f_b1       = (const float*)d_in[11];
  const float* f_w2       = (const float*)d_in[12];
  const float* f_b2       = (const float*)d_in[13];
  const float* gru_wih    = (const float*)d_in[14];
  const float* gru_whh    = (const float*)d_in[15];
  const float* gru_bih    = (const float*)d_in[16];
  const float* gru_bhh    = (const float*)d_in[17];
  const float* mlp_w1     = (const float*)d_in[18];
  const float* mlp_b1     = (const float*)d_in[19];
  const float* mlp_w2     = (const float*)d_in[20];
  const float* mlp_b2     = (const float*)d_in[21];
  const float* ln_s_g     = (const float*)d_in[22];
  const float* ln_s_b     = (const float*)d_in[23];
  const float* ln_pf_g    = (const float*)d_in[24];
  const float* ln_pf_b    = (const float*)d_in[25];

  char* ws = (char*)d_ws;
  size_t off = 0;
  auto alloc = [&](size_t bytes) {
    void* p = ws + off;
    off = (off + bytes + 255) & ~(size_t)255;
    return p;
  };
  _Float16* in16  = (_Float16*)alloc(8u * 4096u * 128u * 2u);   // 8 MB
  _Float16* tok16 = (_Float16*)alloc(8u * 4096u * 128u * 2u);
  _Float16* tov16 = (_Float16*)alloc(8u * 4096u * 128u * 2u);
  _Float16* wk16  = (_Float16*)alloc(128u * 128u * 2u);
  _Float16* wv16  = (_Float16*)alloc(128u * 128u * 2u);
  _Float16* w116  = (_Float16*)alloc(128u * 128u * 2u);
  _Float16* w216  = (_Float16*)alloc(128u * 128u * 2u);
  float* dotsb    = (float*)alloc(64u * 4096u * 4u);            // 1 MB
  float* qb       = (float*)alloc(64u * 128u * 4u);
  float* slotsb   = (float*)alloc(64u * 128u * 4u);
  float* updatesb = (float*)alloc(64u * 128u * 4u);
  float* spA      = (float*)alloc(64u * 2u * 4u);
  float* spB      = (float*)alloc(64u * 2u * 4u);

  // --- prep: f32->f16 conversions, seed state ---
  cvt_f32_f16_kernel<<<(4194304 + 255) / 256, 256, 0, stream>>>(inputs, in16, 4194304);
  cvt_f32_f16_kernel<<<64, 256, 0, stream>>>(Wk, wk16, 16384);
  cvt_f32_f16_kernel<<<64, 256, 0, stream>>>(Wv, wv16, 16384);
  cvt_f32_f16_kernel<<<64, 256, 0, stream>>>(f_w1, w116, 16384);
  cvt_f32_f16_kernel<<<64, 256, 0, stream>>>(f_w2, w216, 16384);
  copy_f32_kernel<<<32, 256, 0, stream>>>(slots_init, slotsb, 8192);
  copy_f32_kernel<<<1, 128, 0, stream>>>(S_p0, spA, 128);

  // --- to_k / to_v (WMMA GEMM, f16 output kept L2-resident) ---
  tokv_kernel<<<256, 256, 98304, stream>>>(in16, wk16, wv16, tok16, tov16);

  float* spc = spA;  // S_p used by pos this iteration
  float* spn = spB;  // S_p written this iteration
  for (int iter = 0; iter < 3; ++iter) {
    lnq_kernel<<<64, 128, 0, stream>>>(slotsb, ln_s_g, ln_s_b, Wq, qb);
    dots_kernel<<<dim3(32, 8, 8), 256, 131072, stream>>>(
        tok16, w116, w216, f_b1, f_b2, g_w, g_b, spc, qb, dotsb);
    softmax_slots_kernel<<<128, 256, 0, stream>>>(dotsb);
    attn_norm_kernel<<<dim3(8, 8), 256, 0, stream>>>(dotsb, spn);
    updates_kernel<<<dim3(8, 8), 256, 135168, stream>>>(
        tov16, w116, w216, f_b1, f_b2, g_w, g_b, spc, dotsb, updatesb);
    slot_update_kernel<<<64, 128, 0, stream>>>(
        updatesb, slotsb, gru_wih, gru_whh, gru_bih, gru_bhh,
        mlp_w1, mlp_b1, mlp_w2, mlp_b2, ln_pf_g, ln_pf_b);
    float* tmp = spc; spc = spn; spn = tmp;
  }
  // after 3 swaps, spc points at the S_p written in the last iteration
  pack_out_kernel<<<(8704 + 255) / 256, 256, 0, stream>>>(slotsb, spc, S_r, S_s,
                                                          (float*)d_out);
}
